// IndustrialMAC_HeteroGNN_60430189854962
// MI455X (gfx1250) — matrix operand order repaired and checked by reading, not compiled, not run on hardware
//
#include <hip/hip_runtime.h>
#include <hip/hip_bf16.h>
#include <math.h>

typedef __attribute__((ext_vector_type(16))) _Float16 v16h;
typedef __attribute__((ext_vector_type(8)))  _Float16 v8h;
typedef __attribute__((ext_vector_type(8)))  float    v8f;

#define N_DEVS  8192
#define N_AP    4
#define E_DD_N  262144
#define E_DA_N  32768
#define E_AD_N  32768
#define CHUNK   4096      // samples per conv->fc chunk (32768 total samples)

// ---------------------------------------------------------------------------
// Generic utility kernels
// ---------------------------------------------------------------------------
__global__ void zero_f32_kernel(float* p, int n) {
    int i = blockIdx.x * blockDim.x + threadIdx.x;
    if (i < n) p[i] = 0.0f;
}

// W [K,N] f32 row-major  ->  out [N,Kpad] f16 (transposed, zero padded K)
__global__ void wtrans_kernel(const float* __restrict__ W, int K, int N, int Kpad,
                              _Float16* __restrict__ out) {
    int idx = blockIdx.x * blockDim.x + threadIdx.x;
    if (idx >= N * Kpad) return;
    int n = idx / Kpad, k = idx - n * Kpad;
    out[idx] = (k < K) ? (_Float16)W[(size_t)k * N + n] : (_Float16)0.0f;
}

__global__ void deg_count_kernel(const int* __restrict__ idx, int E, float* deg) {
    int i = blockIdx.x * blockDim.x + threadIdx.x;
    if (i < E) atomicAdd(&deg[idx[i]], 1.0f);
}

__global__ void finalize_rs_kernel(float* deg, int n) {
    int i = blockIdx.x * blockDim.x + threadIdx.x;
    if (i < n) deg[i] = rsqrtf(fmaxf(deg[i], 1.0f));
}

// agg[dst] += x[src] * rs_out[src]   (64 features per edge)
__global__ void gather_scatter_kernel(const float* __restrict__ x,
                                      const int* __restrict__ src,
                                      const int* __restrict__ dst,
                                      const float* __restrict__ rs_out,
                                      int E, float* __restrict__ agg) {
    int idx = blockIdx.x * blockDim.x + threadIdx.x;
    if (idx >= E * 64) return;
    int e = idx >> 6, f = idx & 63;
    int s = src[e], d = dst[e];
    atomicAdd(&agg[(size_t)d * 64 + f], x[(size_t)s * 64 + f] * rs_out[s]);
}

// out_f16 = (f16)(agg * rs_in[row])   for [n,64]
__global__ void scale_f16_kernel(const float* __restrict__ agg,
                                 const float* __restrict__ rs_in,
                                 int n, _Float16* __restrict__ out) {
    int idx = blockIdx.x * blockDim.x + threadIdx.x;
    if (idx >= n * 64) return;
    out[idx] = (_Float16)(agg[idx] * rs_in[idx >> 6]);
}

// ---------------------------------------------------------------------------
// WMMA GEMM:  out = act( A[M,K](f16) x Bt[N,K]^T(f16) + bias[N] (+prev) )
// Block covers 64 rows (4 x 16 M-tiles) x N cols. One wave per 16-col slice;
// each wave computes 4 16x16 tiles, reusing each B fragment 4x.
// blockDim.x = 2*N (= 32 * N/16 waves); grid.x = ceil(M/64).
// K must be a multiple of 32. A buffer must be readable for 64-row blocks
// (rows >= M are loaded but never stored: WMMA rows are independent).
// ---------------------------------------------------------------------------
__global__ void gemm16_wmma_kernel(const _Float16* __restrict__ A,
                                   const _Float16* __restrict__ Bt,
                                   const float* __restrict__ bias,
                                   float* __restrict__ outF32,
                                   _Float16* __restrict__ outF16,
                                   int f16_ld, int f16_off,
                                   int M, int K, int N, int relu, int addprev) {
    int lane = threadIdx.x & 31;
    int wave = threadIdx.x >> 5;
    int m0   = blockIdx.x * 64;
    int colg = wave * 16 + (lane & 15);
    const int kbA = (lane < 16) ? 0 : 8;    // A frag: K 0-7 & 16-23 | 8-15 & 24-31
    const int kbB = (lane < 16) ? 0 : 8;    // B frag: K 0-7 & 16-23 | 8-15 & 24-31

    const _Float16* Bp = Bt + (size_t)colg * K;
    const _Float16* Ap0 = A + (size_t)(m0 +  0 + (lane & 15)) * K;
    const _Float16* Ap1 = A + (size_t)(m0 + 16 + (lane & 15)) * K;
    const _Float16* Ap2 = A + (size_t)(m0 + 32 + (lane & 15)) * K;
    const _Float16* Ap3 = A + (size_t)(m0 + 48 + (lane & 15)) * K;

    v8f acc0 = {}, acc1 = {}, acc2 = {}, acc3 = {};
    for (int k0 = 0; k0 < K; k0 += 32) {
        v8h b0 = *(const v8h*)(Bp + k0 + kbB);
        v8h b1 = *(const v8h*)(Bp + k0 + 16 + kbB);
        v16h bf = __builtin_shufflevector(b0, b1, 0,1,2,3,4,5,6,7,8,9,10,11,12,13,14,15);

        v8h x0, x1;
        x0 = *(const v8h*)(Ap0 + k0 + kbA);
        x1 = *(const v8h*)(Ap0 + k0 + 16 + kbA);
        v16h af0 = __builtin_shufflevector(x0, x1, 0,1,2,3,4,5,6,7,8,9,10,11,12,13,14,15);
        x0 = *(const v8h*)(Ap1 + k0 + kbA);
        x1 = *(const v8h*)(Ap1 + k0 + 16 + kbA);
        v16h af1 = __builtin_shufflevector(x0, x1, 0,1,2,3,4,5,6,7,8,9,10,11,12,13,14,15);
        x0 = *(const v8h*)(Ap2 + k0 + kbA);
        x1 = *(const v8h*)(Ap2 + k0 + 16 + kbA);
        v16h af2 = __builtin_shufflevector(x0, x1, 0,1,2,3,4,5,6,7,8,9,10,11,12,13,14,15);
        x0 = *(const v8h*)(Ap3 + k0 + kbA);
        x1 = *(const v8h*)(Ap3 + k0 + 16 + kbA);
        v16h af3 = __builtin_shufflevector(x0, x1, 0,1,2,3,4,5,6,7,8,9,10,11,12,13,14,15);

        acc0 = __builtin_amdgcn_wmma_f32_16x16x32_f16(false, af0, false, bf, (short)0, acc0, false, false);
        acc1 = __builtin_amdgcn_wmma_f32_16x16x32_f16(false, af1, false, bf, (short)0, acc1, false, false);
        acc2 = __builtin_amdgcn_wmma_f32_16x16x32_f16(false, af2, false, bf, (short)0, acc2, false, false);
        acc3 = __builtin_amdgcn_wmma_f32_16x16x32_f16(false, af3, false, bf, (short)0, acc3, false, false);
    }

    float bcol = bias[colg];
    v8f accs[4] = {acc0, acc1, acc2, acc3};
#pragma unroll
    for (int mt = 0; mt < 4; ++mt) {
        int rbase = m0 + mt * 16 + ((lane < 16) ? 0 : 8);
#pragma unroll
        for (int r = 0; r < 8; ++r) {
            int row = rbase + r;
            if (row >= M) continue;
            float v = accs[mt][r] + bcol;
            if (addprev) v += outF32[(size_t)row * N + colg];
            if (relu) v = fmaxf(v, 0.0f);
            if (outF32) outF32[(size_t)row * N + colg] = v;
            if (outF16) outF16[(size_t)row * f16_ld + f16_off + colg] = (_Float16)v;
        }
    }
}

// ---------------------------------------------------------------------------
// CSI conv stack: per-sample conv1(2->16,3x3,SAME)+ReLU, conv2(16->32)+ReLU,
// write flattened activation [2048] as f16 (GEMM A layout). One block/sample.
// ---------------------------------------------------------------------------
__global__ void __launch_bounds__(256)
csi_conv_kernel(const float* __restrict__ csi,
                const float* __restrict__ W1, const float* __restrict__ B1,
                const float* __restrict__ W2, const float* __restrict__ B2,
                _Float16* __restrict__ act2, int s0) {
    int s = s0 + blockIdx.x;
    int tid = threadIdx.x;
    __shared__ float sW1[288], sB1[16], sW2[4608], sB2[32];
    __shared__ float sx[2 * 100];     // [2][10][10] zero-padded halo
    __shared__ float s1[16 * 100];    // [16][10][10] zero-padded halo

    for (int i = tid; i < 288;  i += 256) sW1[i] = W1[i];
    for (int i = tid; i < 4608; i += 256) sW2[i] = W2[i];
    if (tid < 16) sB1[tid] = B1[tid];
    if (tid < 32) sB2[tid] = B2[tid];
    for (int i = tid; i < 200;  i += 256) sx[i] = 0.0f;
    for (int i = tid; i < 1600; i += 256) s1[i] = 0.0f;
    __syncthreads();

    const float* in = csi + (size_t)s * 128;   // [F=8][T=8][2] contiguous
    for (int i = tid; i < 128; i += 256) {
        int c = i & 1, w = (i >> 1) & 7, h = i >> 4;
        sx[c * 100 + (h + 1) * 10 + (w + 1)] = in[i];
    }
    __syncthreads();

    // conv1: 16 x 8 x 8 outputs
    for (int o = tid; o < 1024; o += 256) {
        int oc = o >> 6, h = (o >> 3) & 7, w = o & 7;
        float acc = sB1[oc];
        for (int ic = 0; ic < 2; ++ic) {
            const float* wp = sW1 + oc * 18 + ic * 9;
            const float* xp = sx + ic * 100 + h * 10 + w;
            acc += wp[0]*xp[0]  + wp[1]*xp[1]  + wp[2]*xp[2]
                 + wp[3]*xp[10] + wp[4]*xp[11] + wp[5]*xp[12]
                 + wp[6]*xp[20] + wp[7]*xp[21] + wp[8]*xp[22];
        }
        s1[oc * 100 + (h + 1) * 10 + (w + 1)] = fmaxf(acc, 0.0f);
    }
    __syncthreads();

    // conv2: 32 x 8 x 8, flatten k = oc*64 + h*8 + w, store f16
    _Float16* out = act2 + (size_t)blockIdx.x * 2048;
    for (int o = tid; o < 2048; o += 256) {
        int oc = o >> 6, h = (o >> 3) & 7, w = o & 7;
        float acc = sB2[oc];
        for (int ic = 0; ic < 16; ++ic) {
            const float* wp = sW2 + oc * 144 + ic * 9;
            const float* xp = s1 + ic * 100 + h * 10 + w;
            acc += wp[0]*xp[0]  + wp[1]*xp[1]  + wp[2]*xp[2]
                 + wp[3]*xp[10] + wp[4]*xp[11] + wp[5]*xp[12]
                 + wp[6]*xp[20] + wp[7]*xp[21] + wp[8]*xp[22];
        }
        out[o] = (_Float16)fmaxf(acc, 0.0f);
    }
}

// pos_f = relu([x,y,packets] @ W[3,16] + b) -> merged cols 0..15; zero cols 80..95
__global__ void pos_encode_kernel(const float* __restrict__ pos,
                                  const int* __restrict__ pkts,
                                  const float* __restrict__ W,
                                  const float* __restrict__ b,
                                  _Float16* __restrict__ merged) {
    int idx = blockIdx.x * blockDim.x + threadIdx.x;
    if (idx >= N_DEVS * 16) return;
    int n = idx >> 4, j = idx & 15;
    float v = pos[n * 2] * W[j] + pos[n * 2 + 1] * W[16 + j]
            + (float)pkts[n] * W[32 + j] + b[j];
    merged[(size_t)n * 96 + j]      = (_Float16)fmaxf(v, 0.0f);
    merged[(size_t)n * 96 + 80 + j] = (_Float16)0.0f;
}

// h_ap = relu(ap_pos[4,2] @ W[2,64] + b)
__global__ void ap_encode_kernel(const float* __restrict__ ap_pos,
                                 const float* __restrict__ W,
                                 const float* __restrict__ b,
                                 float* __restrict__ h_ap) {
    int idx = threadIdx.x;
    if (idx >= N_AP * 64) return;
    int a = idx >> 6, j = idx & 63;
    float v = ap_pos[a * 2] * W[j] + ap_pos[a * 2 + 1] * W[64 + j] + b[j];
    h_ap[idx] = fmaxf(v, 0.0f);
}

// ---------------------------------------------------------------------------
// Head: per (a,t) row of 8192 logits -> softmax sum into sched_soft,
// first-argmax one-hot into sched_hard. logits [N,32], col = t*4+a.
// ---------------------------------------------------------------------------
__global__ void head_kernel(const float* __restrict__ logits,
                            float* __restrict__ hard,
                            float* __restrict__ soft) {
    const int N = N_DEVS;
    int a = blockIdx.x & 3, t = blockIdx.x >> 2;
    int col = t * 4 + a;
    int tid = threadIdx.x;
    __shared__ float smax[256];
    __shared__ int   sidx[256];
    __shared__ float ssum[256];

    float m = -INFINITY; int mi = 0;
    for (int n = tid; n < N; n += 256) {
        float v = logits[(size_t)n * 32 + col];
        if (v > m) { m = v; mi = n; }
    }
    smax[tid] = m; sidx[tid] = mi;
    __syncthreads();
    for (int s = 128; s > 0; s >>= 1) {
        if (tid < s) {
            if (smax[tid + s] > smax[tid] ||
                (smax[tid + s] == smax[tid] && sidx[tid + s] < sidx[tid])) {
                smax[tid] = smax[tid + s]; sidx[tid] = sidx[tid + s];
            }
        }
        __syncthreads();
    }
    float gmax = smax[0]; int gidx = sidx[0];

    float sum = 0.0f;
    for (int n = tid; n < N; n += 256)
        sum += expf(logits[(size_t)n * 32 + col] - gmax);
    ssum[tid] = sum;
    __syncthreads();
    for (int s = 128; s > 0; s >>= 1) {
        if (tid < s) ssum[tid] += ssum[tid + s];
        __syncthreads();
    }
    float inv = 1.0f / ssum[0];

    for (int n = tid; n < N; n += 256) {
        float p = expf(logits[(size_t)n * 32 + col] - gmax) * inv;
        atomicAdd(&soft[(size_t)n * 8 + t], p);
    }
    if (tid == 0) hard[(size_t)gidx * 8 + t] = 1.0f;
}

// ---------------------------------------------------------------------------
// Launcher
// ---------------------------------------------------------------------------
static inline int cdiv(int a, int b) { return (a + b - 1) / b; }

extern "C" void kernel_launch(void* const* d_in, const int* in_sizes, int n_in,
                              void* d_out, int out_size, void* d_ws, size_t ws_size,
                              hipStream_t stream) {
    (void)in_sizes; (void)n_in; (void)out_size; (void)ws_size;
    const float* device_pos = (const float*)d_in[0];
    const float* ap_pos     = (const float*)d_in[1];
    const float* csi        = (const float*)d_in[2];
    const int*   pkts       = (const int*)d_in[3];
    const int*   dd_src     = (const int*)d_in[4];
    const int*   dd_dst     = (const int*)d_in[5];
    const int*   da_src     = (const int*)d_in[6];
    const int*   da_dst     = (const int*)d_in[7];
    const int*   ad_src     = (const int*)d_in[8];
    const int*   ad_dst     = (const int*)d_in[9];
    const float* cnn1_W = (const float*)d_in[10]; const float* cnn1_b = (const float*)d_in[11];
    const float* cnn2_W = (const float*)d_in[12]; const float* cnn2_b = (const float*)d_in[13];
    const float* csi_fc_W = (const float*)d_in[14]; const float* csi_fc_b = (const float*)d_in[15];
    const float* dev_pos_W = (const float*)d_in[16]; const float* dev_pos_b = (const float*)d_in[17];
    const float* dev_csi_W = (const float*)d_in[18]; const float* dev_csi_b = (const float*)d_in[19];
    const float* dev_merge_W = (const float*)d_in[20]; const float* dev_merge_b = (const float*)d_in[21];
    const float* ap_enc_W = (const float*)d_in[22]; const float* ap_enc_b = (const float*)d_in[23];
    const float* g1_dd_W = (const float*)d_in[24]; const float* g1_dd_b = (const float*)d_in[25];
    const float* g1_da_W = (const float*)d_in[26]; const float* g1_da_b = (const float*)d_in[27];
    const float* g1_ad_W = (const float*)d_in[28]; const float* g1_ad_b = (const float*)d_in[29];
    const float* g2_dd_W = (const float*)d_in[30]; const float* g2_dd_b = (const float*)d_in[31];
    const float* g2_ad_W = (const float*)d_in[34]; const float* g2_ad_b = (const float*)d_in[35];
    const float* aph1_W = (const float*)d_in[36]; const float* aph1_b = (const float*)d_in[37];
    const float* aph2_W = (const float*)d_in[38]; const float* aph2_b = (const float*)d_in[39];

    float* out = (float*)d_out;
    float* out_hard   = out;                 // [8192, 8]
    float* out_logits = out + 65536;         // [8192, 8, 4]
    float* out_soft   = out + 65536 + 262144;// [8192, 8]

    // ---- workspace layout ----
    char* ws = (char*)d_ws;
    size_t off = 0;
    auto alloc = [&](size_t bytes) { size_t o = off; off = (off + bytes + 255) & ~(size_t)255; return o; };
    _Float16* act2       = (_Float16*)(ws + alloc((size_t)CHUNK * 2048 * 2));   // per-chunk conv out
    _Float16* csi_emb16  = (_Float16*)(ws + alloc((size_t)N_DEVS * 256 * 2));   // [8192,256]
    _Float16* merged16   = (_Float16*)(ws + alloc((size_t)N_DEVS * 96 * 2));    // [8192,96]
    float*    h_dev      = (float*)   (ws + alloc((size_t)N_DEVS * 64 * 4));
    float*    h_ap       = (float*)   (ws + alloc((size_t)N_AP * 64 * 4));
    float*    hd1        = (float*)   (ws + alloc((size_t)N_DEVS * 64 * 4));
    float*    ha1        = (float*)   (ws + alloc((size_t)64 * 64 * 4));        // 64-row padded
    float*    hd2        = (float*)   (ws + alloc((size_t)N_DEVS * 64 * 4));
    _Float16* hd2_16     = (_Float16*)(ws + alloc((size_t)N_DEVS * 64 * 2));
    _Float16* t1_16      = (_Float16*)(ws + alloc((size_t)N_DEVS * 64 * 2));
    float*    aggA       = (float*)   (ws + alloc((size_t)N_DEVS * 64 * 4));
    _Float16* agg16      = (_Float16*)(ws + alloc((size_t)N_DEVS * 64 * 2));
    float*    rs         = (float*)   (ws + alloc((size_t)32776 * 4));
    float* rs_dd_out = rs;            // 8192
    float* rs_dd_in  = rs + 8192;     // 8192
    float* rs_da_out = rs + 16384;    // 8192
    float* rs_da_in  = rs + 24576;    // 4
    float* rs_ad_out = rs + 24580;    // 4
    float* rs_ad_in  = rs + 24584;    // 8192
    _Float16* csi_fcT   = (_Float16*)(ws + alloc((size_t)64 * 2048 * 2));
    _Float16* dev_csiT  = (_Float16*)(ws + alloc((size_t)64 * 256 * 2));
    _Float16* dev_mrgT  = (_Float16*)(ws + alloc((size_t)64 * 96 * 2));
    _Float16* g1ddT     = (_Float16*)(ws + alloc((size_t)64 * 64 * 2));
    _Float16* g1adT     = (_Float16*)(ws + alloc((size_t)64 * 64 * 2));
    _Float16* g1daT     = (_Float16*)(ws + alloc((size_t)64 * 64 * 2));
    _Float16* g2ddT     = (_Float16*)(ws + alloc((size_t)64 * 64 * 2));
    _Float16* g2adT     = (_Float16*)(ws + alloc((size_t)64 * 64 * 2));
    _Float16* aph1T     = (_Float16*)(ws + alloc((size_t)64 * 64 * 2));
    _Float16* aph2T     = (_Float16*)(ws + alloc((size_t)32 * 64 * 2));

    // ---- weight transpose/convert to f16 [N,Kpad] ----
    auto wtrans = [&](const float* W, int K, int N, int Kpad, _Float16* dst) {
        wtrans_kernel<<<cdiv(N * Kpad, 256), 256, 0, stream>>>(W, K, N, Kpad, dst);
    };
    wtrans(csi_fc_W, 2048, 64, 2048, csi_fcT);
    wtrans(dev_csi_W, 256, 64, 256, dev_csiT);
    wtrans(dev_merge_W, 80, 64, 96, dev_mrgT);
    wtrans(g1_dd_W, 64, 64, 64, g1ddT);
    wtrans(g1_ad_W, 64, 64, 64, g1adT);
    wtrans(g1_da_W, 64, 64, 64, g1daT);
    wtrans(g2_dd_W, 64, 64, 64, g2ddT);
    wtrans(g2_ad_W, 64, 64, 64, g2adT);
    wtrans(aph1_W, 64, 64, 64, aph1T);
    wtrans(aph2_W, 64, 32, 64, aph2T);

    // ---- degrees -> rsqrt(max(deg,1)) ----
    zero_f32_kernel<<<cdiv(32776, 256), 256, 0, stream>>>(rs, 32776);
    deg_count_kernel<<<cdiv(E_DD_N, 256), 256, 0, stream>>>(dd_src, E_DD_N, rs_dd_out);
    deg_count_kernel<<<cdiv(E_DD_N, 256), 256, 0, stream>>>(dd_dst, E_DD_N, rs_dd_in);
    deg_count_kernel<<<cdiv(E_DA_N, 256), 256, 0, stream>>>(da_src, E_DA_N, rs_da_out);
    deg_count_kernel<<<cdiv(E_DA_N, 256), 256, 0, stream>>>(da_dst, E_DA_N, rs_da_in);
    deg_count_kernel<<<cdiv(E_AD_N, 256), 256, 0, stream>>>(ad_src, E_AD_N, rs_ad_out);
    deg_count_kernel<<<cdiv(E_AD_N, 256), 256, 0, stream>>>(ad_dst, E_AD_N, rs_ad_in);
    finalize_rs_kernel<<<cdiv(32776, 256), 256, 0, stream>>>(rs, 32776);

    // ---- encoders ----
    pos_encode_kernel<<<cdiv(N_DEVS * 16, 256), 256, 0, stream>>>(device_pos, pkts,
                                                                  dev_pos_W, dev_pos_b, merged16);
    ap_encode_kernel<<<1, 256, 0, stream>>>(ap_pos, ap_enc_W, ap_enc_b, h_ap);

    auto gemm = [&](const _Float16* A, const _Float16* Bt, const float* bias,
                    float* of32, _Float16* of16, int ld16, int off16,
                    int M, int K, int N, int relu, int addprev) {
        gemm16_wmma_kernel<<<cdiv(M, 64), 2 * N, 0, stream>>>(A, Bt, bias, of32, of16,
                                                              ld16, off16, M, K, N, relu, addprev);
    };

    // ---- CSI conv + csi_fc GEMM, chunked over 32768 samples ----
    for (int c = 0; c < (N_DEVS * N_AP) / CHUNK; ++c) {
        int s0 = c * CHUNK;
        csi_conv_kernel<<<CHUNK, 256, 0, stream>>>(csi, cnn1_W, cnn1_b, cnn2_W, cnn2_b, act2, s0);
        gemm(act2, csi_fcT, csi_fc_b, nullptr, csi_emb16 + (size_t)s0 * 64, 64, 0,
             CHUNK, 2048, 64, 1, 0);
    }

    // ---- device encoder: csi_f -> merged cols 16..79, then merge -> h_dev ----
    gemm(csi_emb16, dev_csiT, dev_csi_b, nullptr, merged16, 96, 16, N_DEVS, 256, 64, 1, 0);
    gemm(merged16, dev_mrgT, dev_merge_b, h_dev, nullptr, 0, 0, N_DEVS, 96, 64, 1, 0);

    auto graph_conv = [&](const float* x, const int* src, const int* dst,
                          const float* rso, const float* rsi, int E, int n_dst,
                          const _Float16* Wt, const float* b,
                          float* of32, _Float16* of16, int relu, int addprev) {
        zero_f32_kernel<<<cdiv(n_dst * 64, 256), 256, 0, stream>>>(aggA, n_dst * 64);
        gather_scatter_kernel<<<cdiv(E * 64, 256), 256, 0, stream>>>(x, src, dst, rso, E, aggA);
        scale_f16_kernel<<<cdiv(n_dst * 64, 256), 256, 0, stream>>>(aggA, rsi, n_dst, agg16);
        gemm(agg16, Wt, b, of32, of16, 64, 0, n_dst, 64, 64, relu, addprev);
    };

    // ---- hetero layer 1 ----
    graph_conv(h_dev, dd_src, dd_dst, rs_dd_out, rs_dd_in, E_DD_N, N_DEVS,
               g1ddT, g1_dd_b, hd1, nullptr, 0, 0);
    graph_conv(h_ap, ad_src, ad_dst, rs_ad_out, rs_ad_in, E_AD_N, N_DEVS,
               g1adT, g1_ad_b, hd1, nullptr, 1, 1);
    graph_conv(h_dev, da_src, da_dst, rs_da_out, rs_da_in, E_DA_N, N_AP,
               g1daT, g1_da_b, ha1, nullptr, 1, 0);

    // ---- hetero layer 2 ----
    graph_conv(hd1, dd_src, dd_dst, rs_dd_out, rs_dd_in, E_DD_N, N_DEVS,
               g2ddT, g2_dd_b, hd2, nullptr, 0, 0);
    graph_conv(ha1, ad_src, ad_dst, rs_ad_out, rs_ad_in, E_AD_N, N_DEVS,
               g2adT, g2_ad_b, hd2, hd2_16, 1, 1);

    // ---- ap_head ----
    gemm(hd2_16, aph1T, aph1_b, nullptr, t1_16, 64, 0, N_DEVS, 64, 64, 1, 0);
    gemm(t1_16, aph2T, aph2_b, out_logits, nullptr, 0, 0, N_DEVS, 64, 32, 0, 0);

    // ---- eval head: softmax/argmax over devices per (a,t) ----
    zero_f32_kernel<<<cdiv(65536, 256), 256, 0, stream>>>(out_hard, 65536);
    zero_f32_kernel<<<cdiv(65536, 256), 256, 0, stream>>>(out_soft, 65536);
    head_kernel<<<32, 256, 0, stream>>>(out_logits, out_hard, out_soft);
}
